// ScaledDotProductAttention_35235911696408
// MI455X (gfx1250) — compile-verified, matrix operand order checked by reading, not run on hardware
//
#include <hip/hip_runtime.h>

typedef _Float16 f16;
typedef __attribute__((ext_vector_type(16))) _Float16 v16h;
typedef __attribute__((ext_vector_type(8)))  _Float16 v8h;
typedef __attribute__((ext_vector_type(2)))  _Float16 v2h;
typedef __attribute__((ext_vector_type(8)))  float    v8f;

#define S_LEN  2048
#define DHEAD  64
#define NHEAD  16
#define BATCH  2
#define BM     128   // query rows per workgroup
#define WM     16    // query rows per wave
#define BN     32    // key block
#define NWAVE  8
#define KSTR   72    // sK row stride (halves): 64 + 8 pad, 144B (16B mult)
#define VSTR   40    // sVt row stride (halves): 32 + 8 pad, 80B
#define PSTR   40    // sP row stride (halves)

static __device__ __forceinline__ v16h cat8(v8h lo, v8h hi) {
  return __builtin_shufflevector(lo, hi, 0,1,2,3,4,5,6,7,8,9,10,11,12,13,14,15);
}

// VALU-pipe butterfly max step via DPP16 (stays within each 16-lane half)
template <int CTRL>
static __device__ __forceinline__ float dpp_max(float x) {
  int xi = __float_as_int(x);
  int yi = __builtin_amdgcn_update_dpp(xi, xi, CTRL, 0xF, 0xF, true);
  return fmaxf(x, __int_as_float(yi));
}
// full 16-lane max: xor1, xor2, lane^7 (row_half_mirror), lane^15 (row_mirror)
static __device__ __forceinline__ float rowmax16(float x) {
  x = dpp_max<0xB1>(x);   // quad_perm(1,0,3,2)
  x = dpp_max<0x4E>(x);   // quad_perm(2,3,0,1)
  x = dpp_max<0x141>(x);  // row_half_mirror
  x = dpp_max<0x140>(x);  // row_mirror
  return x;
}

__launch_bounds__(256)
__global__ void fa_fwd_causal_kernel(const float* __restrict__ Qg,
                                     const float* __restrict__ Kg,
                                     const float* __restrict__ Vg,
                                     float* __restrict__ Og)
{
  __shared__ __align__(16) f16 sK[BN * KSTR];          // [key][d]  row-major
  __shared__ __align__(16) f16 sV[DHEAD * VSTR];       // [d][slot] transposed, key-permuted
  __shared__ __align__(16) f16 sP[NWAVE * WM * PSTR];  // per-wave P tiles (key-permuted)

  const int tid  = threadIdx.x;
  const int wave = tid >> 5;
  const int lane = tid & 31;
  const int hl   = lane >> 4;   // lane half (0/1)
  const int l16  = lane & 15;

  const int qtiles = S_LEN / BM;           // 16
  const int bh = blockIdx.x / qtiles;      // fused batch*head
  const int qt = blockIdx.x % qtiles;

  const float* Qb = Qg + (size_t)bh * (S_LEN * DHEAD) + (size_t)qt * (BM * DHEAD);
  const float* Kb = Kg + (size_t)bh * (S_LEN * DHEAD);
  const float* Vb = Vg + (size_t)bh * (S_LEN * DHEAD);
  float*       Ob = Og + (size_t)bh * (S_LEN * DHEAD) + (size_t)qt * (BM * DHEAD);

  const int qwave = qt * BM + wave * WM;   // first absolute query row of this wave

  // ---- load Q once into A-operand layout, pre-scaled by log2(e)/sqrt(D) ----
  const float qscale = 0.18033688f;        // log2(e) / 8
  v16h qa[2];
  {
    const float* qrow = Qb + (size_t)(wave * WM + l16) * DHEAD;
#pragma unroll
    for (int kc = 0; kc < 2; ++kc) {
      const int d0 = kc * 32 + hl * 8;     // A layout: {d0..d0+7, d0+16..d0+23}
      alignas(16) float t[16];
      *(float4*)(t + 0)  = *(const float4*)(qrow + d0 + 0);
      *(float4*)(t + 4)  = *(const float4*)(qrow + d0 + 4);
      *(float4*)(t + 8)  = *(const float4*)(qrow + d0 + 16);
      *(float4*)(t + 12) = *(const float4*)(qrow + d0 + 20);
      v16h q;
#pragma unroll
      for (int i = 0; i < 16; ++i) q[i] = (f16)(t[i] * qscale);
      qa[kc] = q;
    }
  }

  // all-ones B operand: rowsum(P) via one extra WMMA (no lane reductions)
  v16h onesb;
#pragma unroll
  for (int i = 0; i < 16; ++i) onesb[i] = (f16)1.0f;

  const v8f vzero = {0.f,0.f,0.f,0.f,0.f,0.f,0.f,0.f};
  v8f oacc[4];
#pragma unroll
  for (int vt = 0; vt < 4; ++vt) oacc[vt] = vzero;
  float mrow[8], lrow[8];
#pragma unroll
  for (int r = 0; r < 8; ++r) { mrow[r] = -1e30f; lrow[r] = 0.f; }

  const int skey  = tid >> 3;        // staging: key 0..31
  const int sd0   = (tid & 7) * 8;   // staging: d {0,8,..,56}
  // shared contraction-dim permutation for P columns and V rows:
  // key k<16 -> slot 2k ; key>=16 -> slot 2(k-16)+1  (P.V invariant)
  const int vslot = (skey < 16) ? (2 * skey) : (2 * (skey - 16) + 1);
  const int pbase = wave * WM * PSTR;
  const int nblocks = qt * 4 + 4;    // causal: only k <= last query of tile

  for (int kb = 0; kb < nblocks; ++kb) {
    const int k0 = kb * BN;

    __syncthreads();
    // ---- cooperative stage: K row-major f16, V transposed+permuted f16 ----
    {
      const float* ksrc = Kb + (size_t)(k0 + skey) * DHEAD + sd0;
      alignas(16) float tk[8];
      *(float4*)(tk + 0) = *(const float4*)(ksrc + 0);
      *(float4*)(tk + 4) = *(const float4*)(ksrc + 4);
      v8h hk;
#pragma unroll
      for (int i = 0; i < 8; ++i) hk[i] = (f16)tk[i];
      *(v8h*)(sK + skey * KSTR + sd0) = hk;           // 16B ds_store

      const float* vsrc = Vb + (size_t)(k0 + skey) * DHEAD + sd0;
      alignas(16) float tv[8];
      *(float4*)(tv + 0) = *(const float4*)(vsrc + 0);
      *(float4*)(tv + 4) = *(const float4*)(vsrc + 4);
#pragma unroll
      for (int i = 0; i < 8; ++i) sV[(sd0 + i) * VSTR + vslot] = (f16)tv[i];
    }
    if (kb + 1 < nblocks) {   // gfx1250 global_prefetch_b8 on next block
      __builtin_prefetch(Kb + (size_t)(k0 + BN + skey) * DHEAD + sd0, 0, 0);
      __builtin_prefetch(Vb + (size_t)(k0 + BN + skey) * DHEAD + sd0, 0, 0);
    }
    __syncthreads();

    // wave-uniform skip of fully-masked key blocks (EXEC stays all-ones)
    if (k0 > qwave + (WM - 1)) continue;

    // ---- issue ALL LDS operand loads up front (DS returns in order):
    //      K operands complete first for QK^T; V operands drain under softmax
    v16h kop[2][2];
#pragma unroll
    for (int nt = 0; nt < 2; ++nt)
#pragma unroll
      for (int kc = 0; kc < 2; ++kc) {
        const f16* base = sK + (nt * 16 + l16) * KSTR + kc * 32 + hl * 8;
        kop[nt][kc] = cat8(*(const v8h*)base, *(const v8h*)(base + 16));
      }
    v16h vb[4];
#pragma unroll
    for (int vt = 0; vt < 4; ++vt) {
      const f16* vbp = sV + (vt * 16 + l16) * VSTR + hl * 8;
      vb[vt] = cat8(*(const v8h*)vbp, *(const v8h*)(vbp + 16));
    }

    // ---- S = (Q*scale) K^T : 16x32 scores in two f32 tiles ----
    v8f s0 = __builtin_amdgcn_wmma_f32_16x16x32_f16(false, qa[0], false, kop[0][0], (short)0, vzero, false, false);
    s0     = __builtin_amdgcn_wmma_f32_16x16x32_f16(false, qa[1], false, kop[0][1], (short)0, s0,    false, false);
    v8f s1 = __builtin_amdgcn_wmma_f32_16x16x32_f16(false, qa[0], false, kop[1][0], (short)0, vzero, false, false);
    s1     = __builtin_amdgcn_wmma_f32_16x16x32_f16(false, qa[1], false, kop[1][1], (short)0, s1,    false, false);

    // ---- causal mask on diagonal blocks ----
    if (k0 + BN - 1 > qwave) {
#pragma unroll
      for (int r = 0; r < 8; ++r) {
        const int q = qwave + r + hl * 8;
        if (k0 + l16      > q) s0[r] = -1e30f;
        if (k0 + 16 + l16 > q) s1[r] = -1e30f;
      }
    }

    // ---- online softmax (log2 domain): DPP max, packed P store ----
    float alpha[8];
#pragma unroll
    for (int r = 0; r < 8; ++r) {
      const float mx   = rowmax16(fmaxf(s0[r], s1[r]));
      const float mnew = fmaxf(mrow[r], mx);
      alpha[r] = __builtin_amdgcn_exp2f(mrow[r] - mnew);
      mrow[r] = mnew;
      const float p0 = __builtin_amdgcn_exp2f(s0[r] - mnew);
      const float p1 = __builtin_amdgcn_exp2f(s1[r] - mnew);
      // packed store: keys (k0+l16, k0+16+l16) -> slots (2*l16, 2*l16+1)
      const int prow = r + hl * 8;               // C-layout row
      v2h pp; pp[0] = (f16)p0; pp[1] = (f16)p1;
      *(v2h*)(sP + pbase + prow * PSTR + 2 * l16) = pp;
    }

    // ---- P back as A-operand (permuted slots are contiguous) ----
    const f16* pb = sP + pbase + l16 * PSTR + hl * 8;
    const v16h pa = cat8(*(const v8h*)pb, *(const v8h*)(pb + 16));

    // ---- row-sum of P via ones-matrix WMMA, then O += P V ----
    const v8f lsum = __builtin_amdgcn_wmma_f32_16x16x32_f16(false, pa, false, onesb, (short)0, vzero, false, false);
#pragma unroll
    for (int vt = 0; vt < 4; ++vt) {
#pragma unroll
      for (int r = 0; r < 8; ++r) oacc[vt][r] *= alpha[r];   // fused to pk_fma with PV
      oacc[vt] = __builtin_amdgcn_wmma_f32_16x16x32_f16(false, pa, false, vb[vt], (short)0, oacc[vt], false, false);
    }
#pragma unroll
    for (int r = 0; r < 8; ++r) lrow[r] = lrow[r] * alpha[r] + lsum[r];
  }

  // ---- epilogue: normalize and store fp32 ----
#pragma unroll
  for (int r = 0; r < 8; ++r) {
    const float inv = 1.0f / lrow[r];
    float* orow = Ob + (size_t)(wave * WM + r + hl * 8) * DHEAD;
#pragma unroll
    for (int vt = 0; vt < 4; ++vt)
      orow[vt * 16 + l16] = oacc[vt][r] * inv;
  }
}

extern "C" void kernel_launch(void* const* d_in, const int* in_sizes, int n_in,
                              void* d_out, int out_size, void* d_ws, size_t ws_size,
                              hipStream_t stream) {
  (void)in_sizes; (void)n_in; (void)out_size; (void)d_ws; (void)ws_size;
  const float* Q = (const float*)d_in[0];
  const float* K = (const float*)d_in[1];
  const float* V = (const float*)d_in[2];
  float* O = (float*)d_out;
  dim3 grid(BATCH * NHEAD * (S_LEN / BM));   // 512 workgroups
  dim3 block(256);                            // 8 wave32 waves
  fa_fwd_causal_kernel<<<grid, block, 0, stream>>>(Q, K, V, O);
}
